// OscillatoryAttention_16286515986767
// MI455X (gfx1250) — compile-verified, matrix operand order checked by reading, not run on hardware
//
#include <hip/hip_runtime.h>
#include <hip/hip_bf16.h>
#include <math.h>
#include <stdint.h>

// ---------------------------------------------------------------------------
// OscillatoryAttention for MI455X (gfx1250), wave32 + WMMA f16 (16x16x32)
// + Tensor Data Mover (async K-tile staging into LDS, TENSORcnt-tracked).
//
// coherence(i,j) = cos(p_i - p_j) = c_i c_j + s_i s_j  (rank-2)  -> folded
// into QK^T by augmenting head dim to 96:
//   Q' = log2e * [ scale*Q, a*c_i, a*s_i, 0.. ]   (f16)
//   K' =         [ K,       c_j,   s_j,   0.. ]   (f16)
// scores(log2 domain) = Q'.K'^T ; softmax uses exp2. Flash-style streaming
// avoids materializing the 1.07GB score tensor.
// ---------------------------------------------------------------------------

typedef __attribute__((ext_vector_type(16))) _Float16 v16h;
typedef __attribute__((ext_vector_type(8)))  float    v8f;
typedef __attribute__((ext_vector_type(4)))  unsigned int v4u;
typedef __attribute__((ext_vector_type(4)))  unsigned int u32x4;
typedef __attribute__((ext_vector_type(8)))  int          i32x8;
typedef __attribute__((ext_vector_type(4)))  int          i32x4;

#define BB   2
#define SS   2048
#define DD   1024
#define HH   16
#define DK   64
#define DKA  96            // augmented (padded) head dim
#define BSX  (BB*SS)       // 4096
#define LOG2E 1.4426950408889634f

union Frag { v16h v; v4u q[2]; };

static __device__ inline v8f wmma_f16(v16h a, v16h b, v8f c) {
  return __builtin_amdgcn_wmma_f32_16x16x32_f16(
      /*neg_a=*/false, a, /*neg_b=*/false, b,
      /*c_mod=*/(short)0, c, /*reuse_a=*/false, /*reuse_b=*/false);
}

// Load one 16x32 f16 A/B fragment. rowPtr points at this lane's row
// (row = lane%16 of the tile). half = lane/16.
// Element e<8 -> K = k0 + half*8 + e ; e>=8 -> K = k0 + 16 + half*8 + (e-8).
static __device__ inline v16h load_frag(const _Float16* rowPtr, int k0, int half) {
  Frag f;
  f.q[0] = *(const v4u*)(rowPtr + k0 + half * 8);
  f.q[1] = *(const v4u*)(rowPtr + k0 + 16 + half * 8);
  return f.v;
}

// ------------------------------- converters --------------------------------
__global__ void f32_to_f16_kernel(const float* __restrict__ in,
                                  _Float16* __restrict__ out, int n) {
  int i = blockIdx.x * blockDim.x + threadIdx.x;
  if (i < n) out[i] = (_Float16)in[i];
}

// in: (K x N) f32 row-major  ->  out: (N x K) f16 row-major
__global__ void transpose_f32_to_f16_kernel(const float* __restrict__ in,
                                            _Float16* __restrict__ out,
                                            int K, int N) {
  int i = blockIdx.x * blockDim.x + threadIdx.x;
  if (i >= K * N) return;
  int n = i / K, k = i % K;
  out[(size_t)n * K + k] = (_Float16)in[(size_t)k * N + n];
}

// ------------------------------- phase -------------------------------------
// phase[bs][h] = x[bs,:] . Wp[:,h] + bp[h]   (tiny: 0.13 GFLOP)
__global__ void phase_kernel(const float* __restrict__ x,
                             const float* __restrict__ Wp,
                             const float* __restrict__ bp,
                             float* __restrict__ phase) {
  int idx = blockIdx.x * blockDim.x + threadIdx.x;
  if (idx >= BSX * HH) return;
  int bs = idx >> 4, h = idx & 15;
  const float* xr = x + (size_t)bs * DD;
  float acc = bp[h];
  for (int k = 0; k < DD; ++k) acc = fmaf(xr[k], Wp[k * HH + h], acc);
  phase[idx] = acc;
}

// Fill augmented columns 64..95 of Qaug/Kaug.
__global__ void fill_aug_kernel(const float* __restrict__ phase,
                                const float* __restrict__ alpha,
                                _Float16* __restrict__ Qaug,
                                _Float16* __restrict__ Kaug) {
  int idx = blockIdx.x * blockDim.x + threadIdx.x;  // (b,h,s)
  if (idx >= BB * HH * SS) return;
  int s  = idx & (SS - 1);
  int bh = idx >> 11;
  int b  = bh >> 4, h = bh & 15;
  float ph = phase[((size_t)(b * SS + s)) * HH + h];
  float c = cosf(ph), sn = sinf(ph);
  float al = alpha[h];
  _Float16* qr = Qaug + ((size_t)bh * SS + s) * DKA;
  _Float16* kr = Kaug + ((size_t)bh * SS + s) * DKA;
  qr[64] = (_Float16)(al * c  * LOG2E);
  qr[65] = (_Float16)(al * sn * LOG2E);
  kr[64] = (_Float16)c;
  kr[65] = (_Float16)sn;
  for (int t = 66; t < DKA; ++t) { qr[t] = (_Float16)0.f; kr[t] = (_Float16)0.f; }
}

// ------------------------------- WMMA GEMM ---------------------------------
// C(M,N) = A(M,K) x Bt(N,K)^T + bias[n].  Block tile 128x128, 8 waves as
// 2(M) x 4(N); each wave computes 64x32 = 4x2 tiles of 16x16.
// mode 0: outF[gm*N+gn] = v                     (f32, final output)
// mode 1: aug layout  outH[(b,h,s)*96 + d] = (f16)(v*mult)   (Q and K)
// mode 3: V^T layout  outH[((b*16+h)*64+d)*S + s] = (f16)v
__global__ void __launch_bounds__(256)
gemm_wmma_kernel(const _Float16* __restrict__ A, const _Float16* __restrict__ Bt,
                 const float* __restrict__ bias, float* __restrict__ outF,
                 _Float16* __restrict__ outH, int M, int N, int K,
                 int mode, float mult) {
  const int lane = threadIdx.x & 31;
  const int wave = threadIdx.x >> 5;
  const int half = lane >> 4;
  const int ln   = lane & 15;
  const int wm   = wave >> 2;          // 0..1
  const int wn   = wave & 3;           // 0..3
  const int mbase = blockIdx.y * 128 + wm * 64;
  const int nbase = blockIdx.x * 128 + wn * 32;

  v8f acc[4][2] = {};

  for (int k0 = 0; k0 < K; k0 += 32) {
    v16h af[4], bf[2];
#pragma unroll
    for (int mi = 0; mi < 4; ++mi)
      af[mi] = load_frag(A + (size_t)(mbase + mi * 16 + ln) * K, k0, half);
#pragma unroll
    for (int ni = 0; ni < 2; ++ni)
      bf[ni] = load_frag(Bt + (size_t)(nbase + ni * 16 + ln) * K, k0, half);
#pragma unroll
    for (int mi = 0; mi < 4; ++mi)
#pragma unroll
      for (int ni = 0; ni < 2; ++ni)
        acc[mi][ni] = wmma_f16(af[mi], bf[ni], acc[mi][ni]);
  }

#pragma unroll
  for (int mi = 0; mi < 4; ++mi) {
#pragma unroll
    for (int ni = 0; ni < 2; ++ni) {
      const int gn = nbase + ni * 16 + ln;
      const float bv = bias[gn];
      const int gm0 = mbase + mi * 16 + 8 * half;
#pragma unroll
      for (int r = 0; r < 8; ++r) {
        const int gm = gm0 + r;                 // C layout: M = r + 8*half
        float v = acc[mi][ni][r] + bv;
        if (mode == 0) {
          outF[(size_t)gm * N + gn] = v;
        } else {
          int b = gm >> 11, s = gm & (SS - 1);  // M axis = b*S + s
          int h = gn >> 6,  d = gn & 63;        // N axis = h*64 + d
          if (mode == 1) {
            outH[(((size_t)(b * HH + h) * SS + s) * DKA) + d] = (_Float16)(v * mult);
          } else { // mode 3: V transposed (B,H,64,S)
            outH[(((size_t)(b * HH + h) * DK + d) * SS) + s] = (_Float16)v;
          }
        }
      }
    }
  }
}

// ------------------------------- Flash attention ---------------------------
// One wave = one 16-row Q tile; 8 waves/block share the K' stream, which is
// staged into LDS (double-buffered) by the Tensor Data Mover issued from
// wave 0 (TDM ignores EXEC, so issue under a wave-uniform branch) and
// consumed via ds_load_b128 fragment reads. Per 32-column block:
// 6 WMMAs (scores over dk'=96) + 4 WMMAs (P.V).
__global__ void __launch_bounds__(256)
flash_kernel(const _Float16* __restrict__ Qaug, const _Float16* __restrict__ Kaug,
             const _Float16* __restrict__ Vt, float* __restrict__ O) {
  __shared__ __align__(16) _Float16 pstage[8][16 * 32];   // per-wave P staging
  __shared__ __align__(16) _Float16 kstage[2][32 * DKA];  // TDM K' double buffer

  const int lane = threadIdx.x & 31;
  const int wave = threadIdx.x >> 5;
  const int half = lane >> 4;
  const int ln   = lane & 15;
  const int h    = blockIdx.y;
  const int b    = blockIdx.z;
  const int row0 = blockIdx.x * 128 + wave * 16;
  const size_t bh = (size_t)(b * HH + h);

  const _Float16* Qb = Qaug + bh * SS * DKA;
  const _Float16* Kb = Kaug + bh * SS * DKA;
  const _Float16* Vb = Vt   + bh * (size_t)DK * SS;

  // Issue a TDM load of K' rows [j0, j0+32) (32 x 96 f16 tile) into
  // kstage[bufn]. Descriptor per CDNA5 ISA ch.10 (D# groups 0/1).
  auto issue_k_tile = [&](int j0, int bufn) {
    unsigned long long gaddr =
        (unsigned long long)(uintptr_t)(Kb + (size_t)j0 * DKA);
    unsigned lds_addr = (unsigned)(uintptr_t)(&kstage[bufn][0]);
    u32x4 g0;
    g0[0] = 1u;                                   // count=1 (valid), user mode
    g0[1] = lds_addr;                             // lds_addr [63:32]
    g0[2] = (unsigned)gaddr;                      // global_addr [95:64]
    g0[3] = ((unsigned)(gaddr >> 32) & 0x01FFFFFFu) | (2u << 30); // type=2
    i32x8 g1;
    g1[0] = (int)(1u << 16);                      // data_size=1 (2 bytes)
    g1[1] = (int)((unsigned)DKA << 16);           // tensor_dim0 = 96
    g1[2] = (int)((unsigned)SS << 16);            // tensor_dim1 = 2048
    g1[3] = (int)((unsigned)DKA << 16);           // tile_dim0 = 96
    g1[4] = 32;                                   // tile_dim1 = 32
    g1[5] = DKA;                                  // tensor_dim0_stride = 96
    g1[6] = 0;
    g1[7] = 0;
    i32x4 z4 = {0, 0, 0, 0};
#if defined(__clang_major__) && (__clang_major__ >= 23)
    i32x8 z8 = {};
    __builtin_amdgcn_tensor_load_to_lds(g0, g1, z4, z4, z8, 0);
#else
    __builtin_amdgcn_tensor_load_to_lds(g0, g1, z4, z4, 0);
#endif
  };

  // Q fragments (16 rows x 96), loaded once.
  v16h qf[3];
#pragma unroll
  for (int f = 0; f < 3; ++f)
    qf[f] = load_frag(Qb + (size_t)(row0 + ln) * DKA, f * 32, half);

  v8f acc[4] = {};
  float mrow[8], lrow[8];
#pragma unroll
  for (int r = 0; r < 8; ++r) { mrow[r] = -1e30f; lrow[r] = 0.f; }

  if (wave == 0) issue_k_tile(0, 0);              // prime the pipeline

  for (int j0 = 0; j0 < SS; j0 += 32) {
    const int bufn = (j0 >> 5) & 1;
    if (wave == 0) __builtin_amdgcn_s_wait_tensorcnt(0);
    __syncthreads();                              // K' tile ready for all waves
    if (wave == 0 && (j0 + 32) < SS)
      issue_k_tile(j0 + 32, bufn ^ 1);            // async prefetch next tile

    const _Float16* Kl = kstage[bufn];
    v8f s0 = {}, s1 = {};
#pragma unroll
    for (int f = 0; f < 3; ++f) {
      v16h b0 = load_frag(Kl + (size_t)(ln)      * DKA, f * 32, half);
      v16h b1 = load_frag(Kl + (size_t)(16 + ln) * DKA, f * 32, half);
      s0 = wmma_f16(qf[f], b0, s0);
      s1 = wmma_f16(qf[f], b1, s1);
    }

    // Prefetch next V block while softmax VALU work runs.
    if (j0 + 32 < SS) {
#pragma unroll
      for (int n2 = 0; n2 < 4; ++n2)
        __builtin_prefetch(Vb + (size_t)(n2 * 16 + ln) * SS + (j0 + 32), 0, 1);
    }

    // Online softmax (log2 domain). Row r lives in lanes sharing lane/16.
    float p0a[8], p1a[8];
#pragma unroll
    for (int r = 0; r < 8; ++r) {
      float v = fmaxf(s0[r], s1[r]);
      v = fmaxf(v, __shfl_xor(v, 1));
      v = fmaxf(v, __shfl_xor(v, 2));
      v = fmaxf(v, __shfl_xor(v, 4));
      v = fmaxf(v, __shfl_xor(v, 8));
      float mn   = fmaxf(mrow[r], v);
      float corr = exp2f(mrow[r] - mn);
      float p0 = exp2f(s0[r] - mn);
      float p1 = exp2f(s1[r] - mn);
      float rs = p0 + p1;
      rs += __shfl_xor(rs, 1);
      rs += __shfl_xor(rs, 2);
      rs += __shfl_xor(rs, 4);
      rs += __shfl_xor(rs, 8);
      lrow[r] = lrow[r] * corr + rs;
      mrow[r] = mn;
#pragma unroll
      for (int n2 = 0; n2 < 4; ++n2) acc[n2][r] *= corr;
      p0a[r] = p0; p1a[r] = p1;
    }

    // C-layout -> A-layout shuffle via LDS (per-wave region; LDS is in-order
    // within a wave, no barrier needed).
    _Float16* pl = pstage[wave];
#pragma unroll
    for (int r = 0; r < 8; ++r) {
      int Mr = r + 8 * half;
      pl[Mr * 32 + ln]      = (_Float16)p0a[r];
      pl[Mr * 32 + 16 + ln] = (_Float16)p1a[r];
    }
    Frag pf;
    pf.q[0] = *(const v4u*)&pl[ln * 32 + half * 8];
    pf.q[1] = *(const v4u*)&pl[ln * 32 + 16 + half * 8];

    // acc += P(16x32) x V(32x64)
#pragma unroll
    for (int n2 = 0; n2 < 4; ++n2) {
      v16h vf = load_frag(Vb + (size_t)(n2 * 16 + ln) * SS, j0, half);
      acc[n2] = wmma_f16(pf.v, vf, acc[n2]);
    }
    __syncthreads();   // all waves done with kstage[bufn] before overwrite
  }

  // Normalize and write O in (B,S,H*dk) f32.
#pragma unroll
  for (int n2 = 0; n2 < 4; ++n2) {
#pragma unroll
    for (int r = 0; r < 8; ++r) {
      int grow = row0 + r + 8 * half;
      float ov = acc[n2][r] / lrow[r];
      O[((size_t)(b * SS + grow)) * DD + h * DK + n2 * 16 + ln] = ov;
    }
  }
}

// ------------------------------- launcher ----------------------------------
extern "C" void kernel_launch(void* const* d_in, const int* in_sizes, int n_in,
                              void* d_out, int out_size, void* d_ws, size_t ws_size,
                              hipStream_t stream) {
  const float* x  = (const float*)d_in[0];
  const float* Wq = (const float*)d_in[1];
  const float* bq = (const float*)d_in[2];
  const float* Wk = (const float*)d_in[3];
  const float* bk = (const float*)d_in[4];
  const float* Wv = (const float*)d_in[5];
  const float* bv = (const float*)d_in[6];
  const float* Wo = (const float*)d_in[7];
  const float* bo = (const float*)d_in[8];
  const float* Wp = (const float*)d_in[9];
  const float* bp = (const float*)d_in[10];
  const float* alpha = (const float*)d_in[11];
  float* out = (float*)d_out;

  // workspace carve-out (~76 MB), 256B aligned chunks
  char* w = (char*)d_ws;
  size_t off = 0;
  auto carve = [&](size_t bytes) {
    char* p = w + off;
    off = (off + bytes + 255) & ~(size_t)255;
    return p;
  };
  _Float16* Xh   = (_Float16*)carve((size_t)BSX * DD * 2);
  _Float16* Wqt  = (_Float16*)carve((size_t)DD * DD * 2);
  _Float16* Wkt  = (_Float16*)carve((size_t)DD * DD * 2);
  _Float16* Wvt  = (_Float16*)carve((size_t)DD * DD * 2);
  _Float16* Wot  = (_Float16*)carve((size_t)DD * DD * 2);
  _Float16* Qaug = (_Float16*)carve((size_t)BB * HH * SS * DKA * 2);
  _Float16* Kaug = (_Float16*)carve((size_t)BB * HH * SS * DKA * 2);
  _Float16* Vt   = (_Float16*)carve((size_t)BB * HH * DK * SS * 2);
  float*    phase= (float*)   carve((size_t)BSX * HH * 4);
  float*    Obuf = (float*)   carve((size_t)BSX * DD * 4);
  _Float16* Oh   = (_Float16*)carve((size_t)BSX * DD * 2);
  (void)ws_size; (void)n_in; (void)in_sizes; (void)out_size;

  const int nXD = BSX * DD;        // 4,194,304
  const int nW  = DD * DD;         // 1,048,576

  // 1) converts
  f32_to_f16_kernel<<<(nXD + 255) / 256, 256, 0, stream>>>(x, Xh, nXD);
  transpose_f32_to_f16_kernel<<<(nW + 255) / 256, 256, 0, stream>>>(Wq, Wqt, DD, DD);
  transpose_f32_to_f16_kernel<<<(nW + 255) / 256, 256, 0, stream>>>(Wk, Wkt, DD, DD);
  transpose_f32_to_f16_kernel<<<(nW + 255) / 256, 256, 0, stream>>>(Wv, Wvt, DD, DD);
  transpose_f32_to_f16_kernel<<<(nW + 255) / 256, 256, 0, stream>>>(Wo, Wot, DD, DD);

  // 2) phase projection (f32, tiny)
  phase_kernel<<<(BSX * HH + 255) / 256, 256, 0, stream>>>(x, Wp, bp, phase);

  // 3) QKV projections via WMMA GEMM
  dim3 ggrid(DD / 128, BSX / 128);
  const float qmult = 0.125f * LOG2E;  // dk^-0.5 = 1/8, log2e folded for exp2
  gemm_wmma_kernel<<<ggrid, 256, 0, stream>>>(Xh, Wqt, bq, nullptr, Qaug,
                                              BSX, DD, DD, 1, qmult);
  gemm_wmma_kernel<<<ggrid, 256, 0, stream>>>(Xh, Wkt, bk, nullptr, Kaug,
                                              BSX, DD, DD, 1, 1.0f);
  gemm_wmma_kernel<<<ggrid, 256, 0, stream>>>(Xh, Wvt, bv, nullptr, Vt,
                                              BSX, DD, DD, 3, 1.0f);

  // 4) cos/sin augmentation columns
  fill_aug_kernel<<<(BB * HH * SS + 255) / 256, 256, 0, stream>>>(phase, alpha,
                                                                  Qaug, Kaug);

  // 5) flash attention (TDM-staged K', WMMA core)
  dim3 fgrid(SS / 128, HH, BB);
  flash_kernel<<<fgrid, 256, 0, stream>>>(Qaug, Kaug, Vt, Obuf);

  // 6) output projection
  f32_to_f16_kernel<<<(nXD + 255) / 256, 256, 0, stream>>>(Obuf, Oh, nXD);
  gemm_wmma_kernel<<<ggrid, 256, 0, stream>>>(Oh, Wot, bo, out, nullptr,
                                              BSX, DD, DD, 0, 1.0f);
}
